// Seq2Seq_16458314678447
// MI455X (gfx1250) — matrix operand order repaired and checked by reading, not compiled, not run on hardware
//
#include <hip/hip_runtime.h>
#include <hip/hip_bf16.h>

typedef __attribute__((ext_vector_type(16))) _Float16 v16h;
typedef __attribute__((ext_vector_type(8)))  float    v8f;

#define B_    1024
#define TIN   168
#define FIN_E 8
#define H_    256
#define TOUT  96
#define G4    1024   // 4*H

union ABf { uint4 u[2]; v16h h; };

__device__ __forceinline__ float sigm(float x) { return 1.0f / (1.0f + __expf(-x)); }
__device__ __forceinline__ float ftanh(float x) {
    x = fminf(fmaxf(x, -15.0f), 15.0f);
    float e = __expf(2.0f * x);
    return (e - 1.0f) / (e + 1.0f);
}

// One-time prep: f32->f16 weight conversion, bias sums, zero h0/c0, seed decoder input.
__global__ void prep_kernel(const float* __restrict__ encWhh, const float* __restrict__ decWhh,
                            const float* __restrict__ ebih, const float* __restrict__ ebhh,
                            const float* __restrict__ dbih, const float* __restrict__ dbhh,
                            const float* __restrict__ inputs,
                            _Float16* __restrict__ encWhh_h, _Float16* __restrict__ decWhh_h,
                            float* __restrict__ bsum_e, float* __restrict__ bsum_d,
                            _Float16* __restrict__ h0, float* __restrict__ c0,
                            float* __restrict__ pred) {
    int tid = blockIdx.x * blockDim.x + threadIdx.x;   // grid covers 1024*256 = 262144
    encWhh_h[tid] = (_Float16)encWhh[tid];
    decWhh_h[tid] = (_Float16)decWhh[tid];
    h0[tid] = (_Float16)0.0f;
    c0[tid] = 0.0f;
    if (tid < G4) {
        bsum_e[tid] = ebih[tid] + ebhh[tid];
        bsum_d[tid] = dbih[tid] + dbhh[tid];
    }
    if (tid < B_) {
        pred[tid] = inputs[(size_t)tid * (TIN * FIN_E) + (TIN - 1) * FIN_E];  // x0 = inputs[:, -1, 0]
    }
}

// One LSTM timestep. grid = (B/16, H/64), block = 128 (4 waves; wave s = gate section s).
// gates = x @ Wih^T + h @ Whh^T + bsum ; then cell update for this block's 16x64 (row,hcol) tile.
template <int FIN>
__launch_bounds__(128)
__global__ void lstm_step(const float* __restrict__ x, int xstride,
                          const float* __restrict__ Wih,            // (1024, FIN) row-major f32
                          const _Float16* __restrict__ Whh,         // (1024, 256) row-major f16
                          const float* __restrict__ bsum,           // (1024)
                          const _Float16* __restrict__ h_in,        // (B, 256) f16
                          _Float16* __restrict__ h_out,             // (B, 256) f16
                          float* __restrict__ c_buf) {              // (B, 256) f32, in-place
    __shared__ _Float16 h_tile[16 * H_];   // 8 KB: 16 batch rows of h
    __shared__ float    x_tile[16 * 8];    // x (or pred) for the 16 rows
    __shared__ float    gbuf[4 * 16 * 64]; // 16 KB: gate exchange between waves

    const int tid = threadIdx.x;
    const int s   = tid >> 5;          // gate section 0..3 (i,f,g,o)
    const int L   = tid & 31;          // lane in wave
    const int sel = L >> 4;            // lane half
    const int lm  = L & 15;
    const int m0  = blockIdx.x * 16;   // batch tile base
    const int n0  = blockIdx.y * 64;   // hidden-column tile base

    // Stage h tile (16 contiguous rows) into LDS: 512 x b128.
    {
        const uint4* src = (const uint4*)(h_in + (size_t)m0 * H_);
        uint4*       dst = (uint4*)h_tile;
        for (int i = tid; i < (16 * H_ / 8); i += 128) dst[i] = src[i];
    }
    // Stage x tile.
    if (FIN == 8) {
        if (tid < 32) {
            int r = tid >> 1, hf = tid & 1;
            const float4 v = *(const float4*)(x + (size_t)(m0 + r) * xstride + hf * 4);
            *(float4*)(&x_tile[r * 8 + hf * 4]) = v;
        }
    } else {
        if (tid < 16) x_tile[tid * 8] = x[(size_t)(m0 + tid) * xstride];
    }
    __syncthreads();

    const int nbase = s * H_ + n0;     // this wave's first gate column

    // C init: bias + x @ Wih^T contribution. C layout: acc[nt][r] = (m = r + sel*8, n = nbase+nt*16+lm)
    v8f acc[4];
#pragma unroll
    for (int nt = 0; nt < 4; ++nt) {
        const int n = nbase + nt * 16 + lm;
        const float base = bsum[n];
#pragma unroll
        for (int r = 0; r < 8; ++r) {
            const int m = r + sel * 8;
            float v = base;
#pragma unroll
            for (int k = 0; k < FIN; ++k)
                v += x_tile[m * 8 + k] * Wih[(size_t)n * FIN + k];
            acc[nt][r] = v;
        }
    }

    // GEMM: h(16x256) @ Whh^T tile -> 4 x (16x16), K swept in 8 chunks of 32.
#pragma unroll
    for (int kk = 0; kk < 8; ++kk) {
        // A fragment from LDS, contiguous K to match B:
        // lane<16: K kk*32+[0..15]; lane>=16: +[16..31]
        ABf a;
        const _Float16* ap = &h_tile[(L & 15) * H_ + kk * 32 + sel * 16];
        a.u[0] = *(const uint4*)(ap);
        a.u[1] = *(const uint4*)(ap + 8);
#pragma unroll
        for (int nt = 0; nt < 4; ++nt) {
            const int n = nbase + nt * 16 + lm;
            // B fragment: lane n holds Whh[n][k..k+15] contiguous (lanes<16: k=kk*32, else kk*32+16)
            ABf bf;
            const uint4* wp = (const uint4*)&Whh[(size_t)n * H_ + kk * 32 + sel * 16];
            bf.u[0] = wp[0];
            bf.u[1] = wp[1];
            acc[nt] = __builtin_amdgcn_wmma_f32_16x16x32_f16(
                false, a.h, false, bf.h, (short)0, acc[nt], false, false);
        }
    }

    // Exchange gates across waves via LDS.
#pragma unroll
    for (int nt = 0; nt < 4; ++nt)
#pragma unroll
        for (int r = 0; r < 8; ++r)
            gbuf[(s * 16 + r + sel * 8) * 64 + nt * 16 + lm] = acc[nt][r];
    __syncthreads();

    // LSTM cell update: 16x64 elements, 8 per thread.
#pragma unroll
    for (int i = 0; i < 8; ++i) {
        const int idx = tid + i * 128;        // 0..1023
        const int m = idx >> 6, j = idx & 63;
        const float gi = gbuf[(0 * 16 + m) * 64 + j];
        const float gf = gbuf[(1 * 16 + m) * 64 + j];
        const float gg = gbuf[(2 * 16 + m) * 64 + j];
        const float go = gbuf[(3 * 16 + m) * 64 + j];
        const size_t gidx = (size_t)(m0 + m) * H_ + (n0 + j);
        const float c_old = c_buf[gidx];
        const float c_new = sigm(gf) * c_old + sigm(gi) * ftanh(gg);
        const float h_new = sigm(go) * ftanh(c_new);
        c_buf[gidx] = c_new;
        h_out[gidx] = (_Float16)h_new;
    }
}

// Decoder FC: pred[b] = h[b,:] . fc_W + fc_b ; writes feedback buffer + output column t.
// block = 256 (8 waves, one row per wave), grid = B/8 = 128.
__global__ void fc_kernel(const _Float16* __restrict__ h, const float* __restrict__ fcW,
                          const float* __restrict__ fcb, float* __restrict__ pred,
                          float* __restrict__ out, int t) {
    const int wave = threadIdx.x >> 5;
    const int L    = threadIdx.x & 31;
    const int b    = blockIdx.x * 8 + wave;
    float v = 0.0f;
#pragma unroll
    for (int j = 0; j < 8; ++j) {
        const int k = L + j * 32;
        v += (float)h[(size_t)b * H_ + k] * fcW[k];
    }
#pragma unroll
    for (int off = 16; off; off >>= 1) v += __shfl_xor(v, off, 32);
    if (L == 0) {
        const float p = v + fcb[0];
        pred[b] = p;
        out[(size_t)b * TOUT + t] = p;
    }
}

extern "C" void kernel_launch(void* const* d_in, const int* in_sizes, int n_in,
                              void* d_out, int out_size, void* d_ws, size_t ws_size,
                              hipStream_t stream) {
    const float* inputs  = (const float*)d_in[0];
    const float* enc_Wih = (const float*)d_in[1];
    const float* enc_Whh = (const float*)d_in[2];
    const float* enc_bih = (const float*)d_in[3];
    const float* enc_bhh = (const float*)d_in[4];
    const float* dec_Wih = (const float*)d_in[5];
    const float* dec_Whh = (const float*)d_in[6];
    const float* dec_bih = (const float*)d_in[7];
    const float* dec_bhh = (const float*)d_in[8];
    const float* fc_W    = (const float*)d_in[9];
    const float* fc_b    = (const float*)d_in[10];
    float* out = (float*)d_out;

    // Workspace carve-up (all 256B-aligned slices).
    char* p = (char*)d_ws;
    auto take = [&](size_t bytes) { char* q = p; p += (bytes + 255) & ~(size_t)255; return q; };
    _Float16* encWhh_h = (_Float16*)take((size_t)G4 * H_ * 2);   // 512 KB
    _Float16* decWhh_h = (_Float16*)take((size_t)G4 * H_ * 2);   // 512 KB
    float*    bsum_e   = (float*)take(G4 * 4);
    float*    bsum_d   = (float*)take(G4 * 4);
    _Float16* hbuf[2];
    hbuf[0] = (_Float16*)take((size_t)B_ * H_ * 2);              // 512 KB
    hbuf[1] = (_Float16*)take((size_t)B_ * H_ * 2);              // 512 KB
    float*    cbuf     = (float*)take((size_t)B_ * H_ * 4);      // 1 MB
    float*    pred     = (float*)take(B_ * 4);

    prep_kernel<<<1024, 256, 0, stream>>>(enc_Whh, dec_Whh, enc_bih, enc_bhh, dec_bih, dec_bhh,
                                          inputs, encWhh_h, decWhh_h, bsum_e, bsum_d,
                                          hbuf[0], cbuf, pred);

    const dim3 grid(B_ / 16, H_ / 64);
    int par = 0;
    for (int t = 0; t < TIN; ++t) {
        lstm_step<FIN_E><<<grid, 128, 0, stream>>>(inputs + (size_t)t * FIN_E, TIN * FIN_E,
                                                   enc_Wih, encWhh_h, bsum_e,
                                                   hbuf[par], hbuf[par ^ 1], cbuf);
        par ^= 1;
    }
    for (int t = 0; t < TOUT; ++t) {
        lstm_step<1><<<grid, 128, 0, stream>>>(pred, 1,
                                               dec_Wih, decWhh_h, bsum_d,
                                               hbuf[par], hbuf[par ^ 1], cbuf);
        par ^= 1;
        fc_kernel<<<B_ / 8, 256, 0, stream>>>(hbuf[par], fc_W, fc_b, pred, out, t);
    }
}
